// R_HGNN_31001073942570
// MI455X (gfx1250) — matrix operand order repaired and checked by reading, not compiled, not run on hardware
//
#include <hip/hip_runtime.h>
#include <hip/hip_bf16.h>

#define RREL 3
#define NN   20000
#define EE   250000
#define HH   4
#define DD   32
#define DDR  32
#define ININ 128     // H*D
#define RELIN0 64
#define NEGS 0.2f

typedef __attribute__((ext_vector_type(16))) __bf16 v16bf;
typedef __attribute__((ext_vector_type(8)))  float  v8f;

__device__ __forceinline__ float leakyf(float x) { return x > 0.f ? x : NEGS * x; }

__device__ __forceinline__ float atomicMaxFloat(float* addr, float value) {
  // IEEE bit trick: works for mixed signs, init must be a very negative float
  if (value >= 0.f)
    return __int_as_float(atomicMax((int*)addr, __float_as_int(value)));
  else
    return __uint_as_float(atomicMin((unsigned int*)addr, __float_as_uint(value)));
}

// ---------------------------------------------------------------------------
// Pre-pack a row-major f32 weight matrix (K x N, ldb = N) into bf16 laid out
// in the contiguous-k B fragment order that this GEMM's A loads also use:
//   P[((kt*NT + nt)*32 + lane)*16 + i]  where
//   lane<16 : n = nt*16+lane,    k = kt*32 + i       (i = 0..15)
//   lane>=16: n = nt*16+lane-16, k = kt*32 + 16 + i
// Batched: b = idx / (K*N).
// ---------------------------------------------------------------------------
__global__ void pack_b_bf16(const float* __restrict__ B, __bf16* __restrict__ P,
                            int K, int N, int ldb,
                            long long strideB, long long strideP, long long total)
{
  long long idx = (long long)blockIdx.x * blockDim.x + threadIdx.x;
  if (idx >= total) return;
  const int KN = K * N;
  long long b  = idx / KN;
  int pe = (int)(idx % KN);
  int NT = N >> 4;
  int i  = pe & 15;
  int l  = (pe >> 4) & 31;
  int nt = (pe >> 9) % NT;
  int kt = (pe >> 9) / NT;
  int k  = kt * 32 + ((l >> 4) << 4) + i;
  int n  = nt * 16 + (l & 15);
  P[b * strideP + pe] = (__bf16)B[b * strideB + (long long)k * ldb + n];
}

// ---------------------------------------------------------------------------
// bf16 WMMA GEMM, pre-packed B, one wave computes a 16 x (NT*16) output strip:
// the A fragment is loaded ONCE per k-step and reused for all NT column tiles
// (8x less A traffic than one-tile-per-wave). grid = (M/16, 1, batch). K%32==0.
// ---------------------------------------------------------------------------
template<int NT>
__global__ __launch_bounds__(32)
void gemm_bf16_wmma(const float* __restrict__ A, const __bf16* __restrict__ Bp,
                    const float* __restrict__ bias, float* __restrict__ C,
                    int K, int lda, int ldc,
                    long long strideA, long long strideBp, long long strideC)
{
  const int lane = threadIdx.x & 31;
  const int half = lane >> 4;
  const int l15  = lane & 15;
  const long long b = blockIdx.z;
  A  += b * strideA;
  Bp += b * strideBp;
  C  += b * strideC;
  const int m0 = blockIdx.x * 16;

  // hoisted bias (per column tile, per lane); no branches in the hot loop
  float bv[NT];
#pragma unroll
  for (int nt = 0; nt < NT; ++nt) bv[nt] = bias ? bias[nt * 16 + l15] : 0.f;

  const float*  arow = A + (long long)(m0 + l15) * lda;
  const __bf16* bbase = Bp + (long long)lane * 16;

  v8f acc[NT];
#pragma unroll
  for (int nt = 0; nt < NT; ++nt) acc[nt] = (v8f){};

  for (int kb = 0; kb < K; kb += 32) {
    // A fragment: lane<16 -> k = kb+{0..15}; lane>=16 -> k = kb+{16..31}
    const float4* a4 = (const float4*)(arow + kb + half * 16);
    float4 x0 = a4[0], x1 = a4[1], x2 = a4[2], x3 = a4[3];
    v16bf af;
    af[0]  = (__bf16)x0.x; af[1]  = (__bf16)x0.y; af[2]  = (__bf16)x0.z; af[3]  = (__bf16)x0.w;
    af[4]  = (__bf16)x1.x; af[5]  = (__bf16)x1.y; af[6]  = (__bf16)x1.z; af[7]  = (__bf16)x1.w;
    af[8]  = (__bf16)x2.x; af[9]  = (__bf16)x2.y; af[10] = (__bf16)x2.z; af[11] = (__bf16)x2.w;
    af[12] = (__bf16)x3.x; af[13] = (__bf16)x3.y; af[14] = (__bf16)x3.z; af[15] = (__bf16)x3.w;

    const __bf16* bk = bbase + ((long long)(kb >> 5) * NT) * (32 * 16);
#pragma unroll
    for (int nt = 0; nt < NT; ++nt) {
      v16bf bf = *(const v16bf*)(bk + (long long)nt * (32 * 16));
      acc[nt] = __builtin_amdgcn_wmma_f32_16x16x32_bf16(false, af, false, bf,
                                                        (short)0, acc[nt], false, false);
    }
  }
#pragma unroll
  for (int nt = 0; nt < NT; ++nt) {
#pragma unroll
    for (int v = 0; v < 8; ++v) {
      int m = m0 + half * 8 + v;
      C[(long long)m * ldc + nt * 16 + l15] = acc[nt][v] + bv[nt];
    }
  }
}

// ---------------------------------------------------------------------------
__global__ void fill_f32(float* __restrict__ p, float v, long long n) {
  long long i = (long long)blockIdx.x * blockDim.x + threadIdx.x;
  if (i < n) p[i] = v;
}

// out[r,o] = sum_i re[r,i] * W[r,i,o] (+ bias[r,o]); tiny matvecs
__global__ void small_matvec(const float* __restrict__ re, const float* __restrict__ W,
                             const float* __restrict__ bias, float* __restrict__ out,
                             int I, int O)
{
  int idx = blockIdx.x * blockDim.x + threadIdx.x;
  if (idx >= RREL * O) return;
  int r = idx / O, o = idx % O;
  const float* rr = re + (long long)r * I;
  const float* ww = W + (long long)r * I * O + o;
  float acc = bias ? bias[(long long)r * O + o] : 0.f;
  for (int i = 0; i < I; ++i) acc += rr[i] * ww[(long long)i * O];
  out[(long long)r * O + o] = acc;
}

// e_dst[r,n,h] = sum_d h*ra[h,:D], e_src = sum_d h*ra[h,D:]
__global__ void node_scores(const float* __restrict__ h, const float* __restrict__ ra,
                            float* __restrict__ eS, float* __restrict__ eD)
{
  int idx = blockIdx.x * blockDim.x + threadIdx.x;
  if (idx >= RREL * NN * HH) return;
  int hh = idx % HH; int rn = idx / HH; int n = rn % NN; int r = rn / NN;
  const float* hp  = h  + (long long)(r * NN + n) * ININ + hh * DD;
  const float* rap = ra + (long long)r * (HH * 2 * DD) + hh * 2 * DD;
  float ed = 0.f, es = 0.f;
#pragma unroll
  for (int d = 0; d < DD; ++d) { float v = hp[d]; ed += v * rap[d]; es += v * rap[DD + d]; }
  eD[idx] = ed; eS[idx] = es;
}

__global__ void edge_score_max(const int* __restrict__ src, const int* __restrict__ dst,
                               const float* __restrict__ eS, const float* __restrict__ eD,
                               float* __restrict__ p, float* __restrict__ mbuf)
{
  long long idx = (long long)blockIdx.x * blockDim.x + threadIdx.x;
  if (idx >= (long long)RREL * EE) return;
  int e = (int)(idx % EE); int r = (int)(idx / EE);
  int s = src[(long long)r * EE + e], d = dst[(long long)r * EE + e];
  const float* esp = eS + (long long)(r * NN + s) * HH;
  const float* edp = eD + (long long)(r * NN + d) * HH;
  float* mp = mbuf + (long long)(r * NN + d) * HH;
#pragma unroll
  for (int hh = 0; hh < HH; ++hh) {
    float v = leakyf(esp[hh] + edp[hh]);
    p[idx * HH + hh] = v;
    atomicMaxFloat(&mp[hh], v);
  }
}

__global__ void edge_exp_sum(const int* __restrict__ dst, float* __restrict__ p,
                             const float* __restrict__ mbuf, float* __restrict__ sbuf)
{
  long long idx = (long long)blockIdx.x * blockDim.x + threadIdx.x;
  if (idx >= (long long)RREL * EE) return;
  int e = (int)(idx % EE); int r = (int)(idx / EE);
  int d = dst[(long long)r * EE + e];
  const float* mp = mbuf + (long long)(r * NN + d) * HH;
  float* sp = sbuf + (long long)(r * NN + d) * HH;
#pragma unroll
  for (int hh = 0; hh < HH; ++hh) {
    float pv = __expf(p[idx * HH + hh] - mp[hh]);
    p[idx * HH + hh] = pv;
    atomicAdd(&sp[hh], pv);
  }
}

__global__ void edge_scatter(const int* __restrict__ src, const int* __restrict__ dst,
                             const float* __restrict__ p, const float* __restrict__ sbuf,
                             const float* __restrict__ h, float* __restrict__ msg)
{
  long long idx = (long long)blockIdx.x * blockDim.x + threadIdx.x;
  if (idx >= (long long)RREL * EE * HH) return;
  int hh = (int)(idx % HH); long long re = idx / HH;
  int e = (int)(re % EE); int r = (int)(re / EE);
  int s = src[(long long)r * EE + e], d = dst[(long long)r * EE + e];
  float a = p[idx] / (sbuf[(long long)(r * NN + d) * HH + hh] + 1e-16f);
  const float4* hp = (const float4*)(h + (long long)(r * NN + s) * ININ + hh * DD);
  float*        mp = msg + (long long)(r * NN + d) * ININ + hh * DD;
#pragma unroll
  for (int q = 0; q < DD / 4; ++q) {
    float4 hv = hp[q];
    atomicAdd(&mp[4 * q + 0], hv.x * a);
    atomicAdd(&mp[4 * q + 1], hv.y * a);
    atomicAdd(&mp[4 * q + 2], hv.z * a);
    atomicAdd(&mp[4 * q + 3], hv.w * a);
  }
}

// msg = relu(msg)*alpha + res*(1-alpha), alpha = sigmoid(resw)
__global__ void post_residual(float* __restrict__ msg, const float* __restrict__ res,
                              const float* __restrict__ resw)
{
  long long idx = (long long)blockIdx.x * blockDim.x + threadIdx.x;
  long long n = (long long)RREL * NN * ININ;
  if (idx >= n) return;
  float alpha = 1.f / (1.f + __expf(-resw[0]));
  float m = msg[idx]; m = m > 0.f ? m : 0.f;
  msg[idx] = m * alpha + res[idx] * (1.f - alpha);
}

// cross-relation attention; feats (R,N,H,D), Wx (R,H,D), out (R,N,H,D)
__global__ void cross_attn(const float* __restrict__ feats, const float* __restrict__ Wx,
                           float* __restrict__ out)
{
  int idx = blockIdx.x * blockDim.x + threadIdx.x;
  if (idx >= NN * HH) return;
  int hh = idx % HH, n = idx / HH;
  float fv[RREL][DD];
#pragma unroll
  for (int r = 0; r < RREL; ++r) {
    const float4* fp = (const float4*)(feats + (long long)(r * NN + n) * ININ + hh * DD);
#pragma unroll
    for (int q = 0; q < DD / 4; ++q) {
      float4 v = fp[q];
      fv[r][4 * q] = v.x; fv[r][4 * q + 1] = v.y; fv[r][4 * q + 2] = v.z; fv[r][4 * q + 3] = v.w;
    }
  }
#pragma unroll
  for (int ro = 0; ro < RREL; ++ro) {
    const float* wv = Wx + ((long long)ro * HH + hh) * DD;
    float sc[RREL];
#pragma unroll
    for (int ri = 0; ri < RREL; ++ri) {
      float dot = 0.f;
#pragma unroll
      for (int d = 0; d < DD; ++d) dot += fv[ri][d] * wv[d];
      sc[ri] = leakyf(dot);
    }
    float mx = fmaxf(sc[0], fmaxf(sc[1], sc[2]));
    float a0 = __expf(sc[0] - mx), a1 = __expf(sc[1] - mx), a2 = __expf(sc[2] - mx);
    float inv = 1.f / (a0 + a1 + a2);
    a0 *= inv; a1 *= inv; a2 *= inv;
    float* op = out + (long long)(ro * NN + n) * ININ + hh * DD;
#pragma unroll
    for (int d = 0; d < DD; ++d)
      op[d] = fv[0][d] * a0 + fv[1][d] * a1 + fv[2][d] * a2;
  }
}

// rp[r,h,d] = sum_c rel[r,h,c] * Wrf[r,h,c,d]
__global__ void rel_proj(const float* __restrict__ rel, const float* __restrict__ Wrf,
                         float* __restrict__ rp)
{
  int idx = blockIdx.x * blockDim.x + threadIdx.x;
  if (idx >= RREL * HH * DD) return;
  int d = idx % DD; int rh = idx / DD; int hh = rh % HH; int r = rh / HH;
  float acc = 0.f;
#pragma unroll
  for (int c = 0; c < DDR; ++c)
    acc += rel[(long long)r * (HH * DDR) + hh * DDR + c] *
           Wrf[(((long long)r * HH + hh) * DDR + c) * DD + d];
  rp[idx] = acc;
}

// out[n,h,d] = sum_r hf[r,n,h,d] * softmax_r(leaky(dot(hf, rp)))
__global__ void relation_fuse(const float* __restrict__ hf, const float* __restrict__ rp,
                              float* __restrict__ out)
{
  int idx = blockIdx.x * blockDim.x + threadIdx.x;
  if (idx >= NN * HH) return;
  int hh = idx % HH, n = idx / HH;
  float fv[RREL][DD];
  float sc[RREL];
#pragma unroll
  for (int r = 0; r < RREL; ++r) {
    const float4* hp = (const float4*)(hf + (long long)(r * NN + n) * ININ + hh * DD);
    const float*  rv = rp + ((long long)r * HH + hh) * DD;
    float dot = 0.f;
#pragma unroll
    for (int q = 0; q < DD / 4; ++q) {
      float4 v = hp[q];
      fv[r][4 * q] = v.x; fv[r][4 * q + 1] = v.y; fv[r][4 * q + 2] = v.z; fv[r][4 * q + 3] = v.w;
      dot += v.x * rv[4 * q] + v.y * rv[4 * q + 1] + v.z * rv[4 * q + 2] + v.w * rv[4 * q + 3];
    }
    sc[r] = leakyf(dot);
  }
  float mx = fmaxf(sc[0], fmaxf(sc[1], sc[2]));
  float a0 = __expf(sc[0] - mx), a1 = __expf(sc[1] - mx), a2 = __expf(sc[2] - mx);
  float inv = 1.f / (a0 + a1 + a2);
  a0 *= inv; a1 *= inv; a2 *= inv;
  float* op = out + (long long)n * ININ + hh * DD;
#pragma unroll
  for (int d = 0; d < DD; ++d)
    op[d] = fv[0][d] * a0 + fv[1][d] * a1 + fv[2][d] * a2;
}

// ---------------------------------------------------------------------------
static inline int cdiv(long long a, int b) { return (int)((a + b - 1) / b); }

extern "C" void kernel_launch(void* const* d_in, const int* in_sizes, int n_in,
                              void* d_out, int out_size, void* d_ws, size_t ws_size,
                              hipStream_t stream)
{
  const float* x        = (const float*)d_in[0];
  const int*   src      = (const int*)d_in[1];
  const int*   dst      = (const int*)d_in[2];
  const float* proj_W   = (const float*)d_in[3];
  const float* proj_b   = (const float*)d_in[4];
  const float* rel_emb0 = (const float*)d_in[5];
  const float* Wn[2]   = {(const float*)d_in[6],  (const float*)d_in[14]};
  const float* Wr[2]   = {(const float*)d_in[7],  (const float*)d_in[15]};
  const float* Wp[2]   = {(const float*)d_in[8],  (const float*)d_in[16]};
  const float* bp[2]   = {(const float*)d_in[9],  (const float*)d_in[17]};
  const float* Wres[2] = {(const float*)d_in[10], (const float*)d_in[18]};
  const float* bres[2] = {(const float*)d_in[11], (const float*)d_in[19]};
  const float* resw[2] = {(const float*)d_in[12], (const float*)d_in[20]};
  const float* Wx[2]   = {(const float*)d_in[13], (const float*)d_in[21]};
  const float* Wnf     = (const float*)d_in[22];
  const float* Wrf     = (const float*)d_in[23];

  const size_t RNHD = (size_t)RREL * NN * ININ;   // 7,680,000 floats
  float* w = (float*)d_ws;
  float* xp   = w; w += RNHD;                     // projected input
  float* hbuf = w; w += RNHD;                     // h (and later hf)
  float* resb = w; w += RNHD;                     // residual branch
  float* msg  = w; w += RNHD;                     // aggregated messages / feats
  float* fA   = w; w += RNHD;                     // layer-0 output
  float* fB   = w; w += RNHD;                     // layer-1 output
  float* pbuf = w; w += (size_t)RREL * EE * HH;   // per-edge scores/probs
  float* mbuf = w; w += (size_t)RREL * NN * HH;   // segment max
  float* sbuf = w; w += (size_t)RREL * NN * HH;   // segment sum
  float* eS   = w; w += (size_t)RREL * NN * HH;
  float* eD   = w; w += (size_t)RREL * NN * HH;
  float* ra   = w; w += (size_t)RREL * HH * 2 * DD;
  float* reA  = w; w += (size_t)RREL * HH * DDR;
  float* reB  = w; w += (size_t)RREL * HH * DDR;
  float* rp   = w; w += (size_t)RREL * HH * DD;
  // packed bf16 weights (fragment layout); 128x128 = 16384 bf16 = 8192 floats
  __bf16* pProj    = (__bf16*)w; w += 8192;
  __bf16* pWnb[2]  = {(__bf16*)w, (__bf16*)(w + 8192)};  w += 2 * 8192;
  __bf16* pWresb[2]= {(__bf16*)w, (__bf16*)(w + 8192)};  w += 2 * 8192;
  __bf16* pWnf     = (__bf16*)w; w += (RREL * HH * DD * DD) / 2;  // 12288 bf16

  const int TB = 256;

  // 0) pack all weight matrices into WMMA-B bf16 fragment layout
  pack_b_bf16<<<cdiv(ININ * ININ, TB), TB, 0, stream>>>(proj_W, pProj, ININ, ININ, ININ, 0, 0, ININ * ININ);
  for (int l = 0; l < 2; ++l) {
    pack_b_bf16<<<cdiv(ININ * ININ, TB), TB, 0, stream>>>(Wn[l],   pWnb[l],   ININ, ININ, ININ, 0, 0, ININ * ININ);
    pack_b_bf16<<<cdiv(ININ * ININ, TB), TB, 0, stream>>>(Wres[l], pWresb[l], ININ, ININ, ININ, 0, 0, ININ * ININ);
  }
  pack_b_bf16<<<cdiv((long long)RREL * HH * DD * DD, TB), TB, 0, stream>>>(
      Wnf, pWnf, DD, DD, DD, (long long)DD * DD, (long long)DD * DD, (long long)RREL * HH * DD * DD);

  // 1) projection: xp = x @ proj_W + proj_b   (M=R*N=60000, K=128, N=128)
  gemm_bf16_wmma<8><<<dim3((RREL * NN) / 16, 1, 1), 32, 0, stream>>>(
      x, pProj, proj_b, xp, ININ, ININ, ININ, 0, 0, 0);

  for (int l = 0; l < 2; ++l) {
    const float* xin   = (l == 0) ? xp : fA;
    float*       xout  = (l == 0) ? fA : fB;
    const float* rein  = (l == 0) ? rel_emb0 : reA;
    float*       reout = (l == 0) ? reA : reB;
    const int    RI    = (l == 0) ? RELIN0 : HH * DDR;

    // h = xin @ Wn  (batched over r; Wn shared)
    gemm_bf16_wmma<8><<<dim3(NN / 16, 1, RREL), 32, 0, stream>>>(
        xin, pWnb[l], nullptr, hbuf, ININ, ININ, ININ,
        (long long)NN * ININ, 0, (long long)NN * ININ);

    // ra[r] = rein[r] @ Wr[r]   (R x (RI x 2HD))
    small_matvec<<<cdiv((long long)RREL * HH * 2 * DD, TB), TB, 0, stream>>>(
        rein, Wr[l], nullptr, ra, RI, HH * 2 * DD);

    node_scores<<<cdiv((long long)RREL * NN * HH, TB), TB, 0, stream>>>(hbuf, ra, eS, eD);

    fill_f32<<<cdiv((long long)RREL * NN * HH, TB), TB, 0, stream>>>(mbuf, -1e30f, (long long)RREL * NN * HH);
    fill_f32<<<cdiv((long long)RREL * NN * HH, TB), TB, 0, stream>>>(sbuf, 0.f, (long long)RREL * NN * HH);
    fill_f32<<<cdiv((long long)RNHD, TB), TB, 0, stream>>>(msg, 0.f, (long long)RNHD);

    edge_score_max<<<cdiv((long long)RREL * EE, TB), TB, 0, stream>>>(src, dst, eS, eD, pbuf, mbuf);
    edge_exp_sum  <<<cdiv((long long)RREL * EE, TB), TB, 0, stream>>>(dst, pbuf, mbuf, sbuf);
    edge_scatter  <<<cdiv((long long)RREL * EE * HH, TB), TB, 0, stream>>>(src, dst, pbuf, sbuf, hbuf, msg);

    // residual branch: resb = xin @ Wres + bres (batched over r; shared weights)
    gemm_bf16_wmma<8><<<dim3(NN / 16, 1, RREL), 32, 0, stream>>>(
        xin, pWresb[l], bres[l], resb, ININ, ININ, ININ,
        (long long)NN * ININ, 0, (long long)NN * ININ);

    post_residual<<<cdiv((long long)RNHD, TB), TB, 0, stream>>>(msg, resb, resw[l]);

    cross_attn<<<cdiv((long long)NN * HH, TB), TB, 0, stream>>>(msg, Wx[l], xout);

    // relemb update: reout = rein @ Wp + bp
    small_matvec<<<cdiv((long long)RREL * HH * DDR, TB), TB, 0, stream>>>(
        rein, Wp[l], bp[l], reout, RI, HH * DDR);
  }

  // RelationFusing: hf[r,:,h,:] = feats[r,:,h,:] @ Wnf[r,h]   (K=32, Ncols=32, batch over h)
  for (int r = 0; r < RREL; ++r) {
    gemm_bf16_wmma<2><<<dim3(NN / 16, 1, HH), 32, 0, stream>>>(
        fB + (size_t)r * NN * ININ, pWnf + (size_t)r * HH * DD * DD, nullptr,
        hbuf + (size_t)r * NN * ININ,
        DD, ININ, ININ,
        (long long)DD, (long long)DD * DD, (long long)DD);
  }

  rel_proj<<<cdiv((long long)RREL * HH * DD, TB), TB, 0, stream>>>(reB, Wrf, rp);

  relation_fuse<<<cdiv((long long)NN * HH, TB), TB, 0, stream>>>(hbuf, rp, (float*)d_out);
}
